// Centroids_20091857011531
// MI455X (gfx1250) — matrix operand order repaired and checked by reading, not compiled, not run on hardware
//
#include <hip/hip_runtime.h>
#include <hip/hip_bf16.h>

// ---------------------------------------------------------------------------
// VQ quantize for MI455X (gfx1250, wave32, WMMA + TDM, double-buffered).
// argmin via split-bf16 WMMA: x = hi + lo (bf16 each),
//   x.c ~= hi.c_hi + lo.c_hi + hi.c_lo (three independent f32 accumulators).
// B tiles staged into LDS by the Tensor Data Mover with pad_enable producing
// the 528B padded row stride needed for conflict-free ds_load_b128 fragments;
// tile kt+1 DMA overlaps tile kt compute (s_wait_tensorcnt 1, in-order TDM).
// ---------------------------------------------------------------------------

typedef __attribute__((ext_vector_type(16))) __bf16   v16bf;
typedef __attribute__((ext_vector_type(8)))  float    v8f;
typedef __attribute__((ext_vector_type(4)))  unsigned v4u;
typedef __attribute__((ext_vector_type(8)))  int      v8i;
typedef __attribute__((ext_vector_type(4)))  int      v4i;

union Frag16 { v16bf v; uint4 q[2]; };

#ifndef __has_builtin
#define __has_builtin(x) 0
#endif
#if __has_builtin(__builtin_amdgcn_tensor_load_to_lds) && \
    __has_builtin(__builtin_amdgcn_s_wait_tensorcnt)
#define USE_TDM 1
#endif

#define BATCH  64
#define FEAT   256
#define KCENT  1024
#define SP     1024                 // H*W = 32*32
#define BP     264                  // padded LDS row stride (bf16 elems) = 528 B

// s = h*32+w  <->  n-within-batch r = w*32+h   (5-bit field swap)
__device__ __forceinline__ int swap5(int s) { return ((s & 31) << 5) | (s >> 5); }

#ifdef USE_TDM
// One TDM descriptor: copy `units` x 8B contiguous from global into LDS,
// inserting a 16B pad after every 512B so rows land at 528B stride.
// 6-arg builtin (amdgpu-toolchain): (v4u g0, v8i g1, v4i, v4i, v8i, i32 cpol)
__device__ __forceinline__ void tdm_copy(unsigned ldsAddr, const void* gsrc,
                                         unsigned units) {
    unsigned long long ga = (unsigned long long)(size_t)gsrc;
    v4u g0;
    g0.x = 1u;                                             // count=1, user desc
    g0.y = ldsAddr;                                        // LDS byte address
    g0.z = (unsigned)(ga & 0xffffffffull);                 // global_addr lo
    g0.w = (unsigned)((ga >> 32) & 0x1ffffffull) | (2u << 30); // addr hi | type=2
    v8i g1;
    g1[0] = (int)((3u << 16)     // data_size = 8 bytes
                | (1u << 20)     // pad_enable
                | (6u << 22)     // pad_interval: every 128 dwords (512B)
                | (3u << 25));   // pad_amount: 4 dwords (16B)
    g1[1] = (int)(units << 16);  // tensor_dim0[15:0]
    g1[2] = (int)(1u << 16);     // tensor_dim0[31:16]=0 | tensor_dim1[15:0]=1
    g1[3] = (int)(units << 16);  // tensor_dim1[31:16]=0 | tile_dim0
    g1[4] = 1;                   // tile_dim1=1, tile_dim2=0
    g1[5] = (int)units;          // tensor_dim0_stride lo32
    g1[6] = 0;
    g1[7] = 0;
    v4i z4 = {0, 0, 0, 0};
    v8i z8 = {0, 0, 0, 0, 0, 0, 0, 0};
    __builtin_amdgcn_tensor_load_to_lds(g0, g1, z4, z4, z8, 0);
}
#endif

// ---------------------------------------------------------------------------
// Prep: centroids f32 -> bf16 hi/lo in per-tile layout + column norms.
// cT layout: tile kt (16 centroids): rows 0-15 = hi, rows 16-31 = lo,
// each row 256 bf16 contiguous;  cT[(kt*32 + r)*256 + f].   grid = 4 x 256
// ---------------------------------------------------------------------------
__global__ __launch_bounds__(256) void kPrep(const float* __restrict__ cent,
                                             __bf16* __restrict__ cT,
                                             float* __restrict__ cnorm) {
    int k = blockIdx.x * 256 + threadIdx.x;          // 0..1023
    int ktile = k >> 4, krow = k & 15;
    __bf16* hiRow = cT + (size_t)(ktile * 32 + krow) * FEAT;
    __bf16* loRow = cT + (size_t)(ktile * 32 + 16 + krow) * FEAT;
    float acc = 0.f;
    for (int f4 = 0; f4 < FEAT; f4 += 4) {
        union { __bf16 b[4]; uint2 q; } ph, pl;
#pragma unroll
        for (int j = 0; j < 4; ++j) {
            float v = cent[(f4 + j) * KCENT + k];
            __bf16 h = (__bf16)v;
            ph.b[j] = h;
            pl.b[j] = (__bf16)(v - (float)h);
            acc += v * v;
        }
        *(uint2*)(hiRow + f4) = ph.q;
        *(uint2*)(loRow + f4) = pl.q;
    }
    cnorm[k] = acc;
}

__global__ __launch_bounds__(256) void kZero(unsigned* __restrict__ counts) {
    for (int i = threadIdx.x; i < KCENT; i += 256) counts[i] = 0u;
}

// ---------------------------------------------------------------------------
// Argmin kernel. 512 WGs x 256 threads (8 waves). Block = 128 rows (s-range),
// wave = 16 rows. Loops 64 column tiles of 16 centroids, double-buffered.
// ---------------------------------------------------------------------------
__global__ __launch_bounds__(256) void kArgmin(const float* __restrict__ x,
                                               const __bf16* __restrict__ cT,
                                               const float* __restrict__ cnorm,
                                               int* __restrict__ indg,
                                               unsigned* __restrict__ counts) {
    __shared__ __align__(16) __bf16 sB[2][32 * BP];  // per buf: rows 0-15 hi, 16-31 lo
    __shared__ float sCn[KCENT];

    const int tid  = threadIdx.x;
    const int wv   = tid >> 5;
    const int ln   = tid & 31;
    const int half = ln >> 4;                // 0: lanes 0-15, 1: lanes 16-31
    const int bb   = blockIdx.x >> 3;        // batch index
    const int s0   = (blockIdx.x & 7) << 7;  // spatial base (128 rows)

    for (int i = tid; i < KCENT; i += 256) sCn[i] = cnorm[i];

#ifdef USE_TDM
    const int      wvu    = __builtin_amdgcn_readfirstlane(wv);
    const unsigned ldsB0  = (unsigned)(size_t)(&sB[0][0]) + (unsigned)(wvu * 4 * 528);
    const unsigned ldsB1  = (unsigned)(size_t)(&sB[1][0]) + (unsigned)(wvu * 4 * 528);
    const char*    gTile0 = (const char*)cT + (size_t)wvu * 2048;
    // prefetch tile 0 into buffer 0 (overlaps the A-fragment loads below)
    tdm_copy(ldsB0, gTile0, 256u);
#endif

    // ---- A fragments (16 rows x 256 feats, bf16 hi/lo) straight from global.
    // A 16-bit 16x32 layout: lane(l&15)=M; K=half*16+e (contiguous, matches B fragments)
    const int    row = (wv << 4) + (ln & 15);           // local row 0..127
    const float* xb  = x + (size_t)bb * (FEAT * SP) + (s0 + row);
    v16bf ahi[8], alo[8];
#pragma unroll
    for (int ks = 0; ks < 8; ++ks) {
        Frag16 fh, fl;
#pragma unroll
        for (int e = 0; e < 16; ++e) {
            int kk = half * 16 + e;
            float v  = xb[(size_t)(ks * 32 + kk) * SP];
            __bf16 h = (__bf16)v;
            fh.v[e] = h;
            fl.v[e] = (__bf16)(v - (float)h);
        }
        ahi[ks] = fh.v;
        alo[ks] = fl.v;
    }

    float best[8];
    int   bestk[8];
#pragma unroll
    for (int v = 0; v < 8; ++v) { best[v] = 3.4e38f; bestk[v] = 0; }

    for (int kt = 0; kt < 64; ++kt) {
#ifdef USE_TDM
        // issue DMA for tile kt+1 into the other buffer (kt=63 wraps: harmless,
        // retired by s_endpgm's implicit wait-idle, buffer never read again)
        {
            int nx = (kt + 1) & 63;
            unsigned dst = ((kt + 1) & 1) ? ldsB1 : ldsB0;
            tdm_copy(dst, gTile0 + (size_t)nx * 16384, 256u);
            __builtin_amdgcn_s_wait_tensorcnt(1);   // in-order => tile kt landed
        }
        __syncthreads();   // all waves' tile-kt chunks visible
#else
        __syncthreads();   // all waves done reading previous tile in this buffer
        {
            const uint4* gsrc = (const uint4*)(cT + (size_t)kt * 32 * FEAT);
            char* dst = (char*)&sB[kt & 1][0];
#pragma unroll
            for (int p = 0; p < 4; ++p) {
                int g = p * 256 + tid;               // 16B granule, 1024 total
                int r = g >> 5, col = g & 31;
                *(uint4*)(dst + r * 528 + col * 16) = gsrc[g];
            }
        }
        __syncthreads();
#endif

        v8f acc0 = {}, acc1 = {}, acc2 = {};   // hi*hi, lo*hi, hi*lo
        const __bf16* sbase = &sB[kt & 1][0];
        const __bf16* bRowH = sbase + (ln & 15) * BP + half * 16;
        const __bf16* bRowL = bRowH + 16 * BP;
#pragma unroll
        for (int ks = 0; ks < 8; ++ks) {
            Frag16 bh, bl;
            bh.q[0] = *(const uint4*)(bRowH + ks * 32);
            bh.q[1] = *(const uint4*)(bRowH + ks * 32 + 8);
            bl.q[0] = *(const uint4*)(bRowL + ks * 32);
            bl.q[1] = *(const uint4*)(bRowL + ks * 32 + 8);
            acc0 = __builtin_amdgcn_wmma_f32_16x16x32_bf16(
                       false, ahi[ks], false, bh.v, (short)0, acc0, false, false);
            acc1 = __builtin_amdgcn_wmma_f32_16x16x32_bf16(
                       false, alo[ks], false, bh.v, (short)0, acc1, false, false);
            acc2 = __builtin_amdgcn_wmma_f32_16x16x32_bf16(
                       false, ahi[ks], false, bl.v, (short)0, acc2, false, false);
        }

        int   kcol = kt * 16 + (ln & 15);
        float cn   = sCn[kcol];
#pragma unroll
        for (int v = 0; v < 8; ++v) {
            float d = cn - 2.0f * (acc0[v] + (acc1[v] + acc2[v]));
            if (d < best[v]) { best[v] = d; bestk[v] = kcol; }
        }

#ifdef USE_TDM
        __syncthreads();   // everyone done reading buf[kt&1] before it is
                           // overwritten by iteration kt+1's DMA issue
#endif
    }

    // ---- reduce across the 16 lanes holding one row (argmin, first-index ties)
#pragma unroll
    for (int v = 0; v < 8; ++v) {
        float d  = best[v];
        int   k2 = bestk[v];
#pragma unroll
        for (int m = 8; m >= 1; m >>= 1) {
            float od = __shfl_xor(d, m, 16);
            int   ok = __shfl_xor(k2, m, 16);
            if (od < d || (od == d && ok < k2)) { d = od; k2 = ok; }
        }
        best[v] = d; bestk[v] = k2;
    }
    if ((ln & 15) == 0) {
#pragma unroll
        for (int v = 0; v < 8; ++v) {
            int r = (wv << 4) + half * 8 + v;    // local row (= C-matrix M)
            int s = s0 + r;
            int n = (bb << 10) + swap5(s);
            indg[n] = bestk[v];
            atomicAdd(&counts[bestk[v]], 1u);
        }
    }
}

// ---------------------------------------------------------------------------
// Segment sums, f-major so x reads stay coalesced. grid=64, WG owns 4 feats.
// ---------------------------------------------------------------------------
__global__ __launch_bounds__(256) void kSums(const float* __restrict__ x,
                                             const int* __restrict__ ind,
                                             float* __restrict__ sums) {
    __shared__ float acc[4 * KCENT];
    __shared__ int   sInd[SP];
    const int tid = threadIdx.x;
    const int f0  = blockIdx.x * 4;
    for (int i = tid; i < 4 * KCENT; i += 256) acc[i] = 0.f;
    for (int b = 0; b < BATCH; ++b) {
        __syncthreads();
        for (int s = tid; s < SP; s += 256) sInd[s] = ind[(b << 10) + swap5(s)];
        __syncthreads();
        for (int cs = 0; cs < 4; ++cs) {
            int s = cs * 256 + tid;
            int k = sInd[s];
#pragma unroll
            for (int ff = 0; ff < 4; ++ff) {
                float v = x[(size_t)b * (FEAT * SP) + (size_t)(f0 + ff) * SP + s];
                atomicAdd(&acc[ff * KCENT + k], v);
            }
        }
    }
    __syncthreads();
    for (int i = tid; i < 4 * KCENT; i += 256)
        sums[(f0 + (i >> 10)) * KCENT + (i & (KCENT - 1))] = acc[i];
}

// ---------------------------------------------------------------------------
// EMA cluster sizes -> cs_norm.  Single workgroup.
// ---------------------------------------------------------------------------
__global__ __launch_bounds__(256) void kNorm(const float* __restrict__ csz,
                                             const unsigned* __restrict__ counts,
                                             float* __restrict__ csn) {
    __shared__ float red[256];
    __shared__ float ncs_s[KCENT];
    const int t = threadIdx.x;
    float part = 0.f;
    for (int i = t; i < KCENT; i += 256) {
        float ncs = 0.99f * csz[i] + 0.01f * (float)counts[i];
        ncs_s[i] = ncs;
        part += ncs;
    }
    red[t] = part;
    __syncthreads();
    for (int o = 128; o > 0; o >>= 1) {
        if (t < o) red[t] += red[t + o];
        __syncthreads();
    }
    float n = red[0];
    const float eps = 1e-5f;
    for (int i = t; i < KCENT; i += 256)
        csn[i] = (ncs_s[i] + eps) / (n + (float)KCENT * eps) * n;
}

// ---------------------------------------------------------------------------
// new_centroids = (0.99*avg + 0.01*sums) / cs_norm[k].  grid=1024
// ---------------------------------------------------------------------------
__global__ __launch_bounds__(256) void kNewC(const float* __restrict__ avg,
                                             const float* __restrict__ sums,
                                             const float* __restrict__ csn,
                                             float* __restrict__ newc) {
    int i = blockIdx.x * 256 + threadIdx.x;
    newc[i] = (0.99f * avg[i] + 0.01f * sums[i]) / csn[i & (KCENT - 1)];
}

// ---------------------------------------------------------------------------
// Gather output in [B,C,H,W] layout.  grid=16384, float4 stores.
// ---------------------------------------------------------------------------
__global__ __launch_bounds__(256) void kGather(const float* __restrict__ newc,
                                               const int* __restrict__ ind,
                                               float* __restrict__ out) {
    int i = blockIdx.x * 256 + threadIdx.x;
    int o = i * 4;
    int b  = o >> 18;
    int c  = (o >> 10) & (FEAT - 1);
    int hw = o & (SP - 1);
    int h  = hw >> 5;
    int w  = hw & 31;
    int nbase = (b << 10) + h;                   // n = b*1024 + w*32 + h
    const float* rowp = newc + c * KCENT;
    float4 r;
    r.x = rowp[ind[nbase + (w + 0) * 32]];
    r.y = rowp[ind[nbase + (w + 1) * 32]];
    r.z = rowp[ind[nbase + (w + 2) * 32]];
    r.w = rowp[ind[nbase + (w + 3) * 32]];
    *(float4*)(out + o) = r;
}

// ---------------------------------------------------------------------------
extern "C" void kernel_launch(void* const* d_in, const int* in_sizes, int n_in,
                              void* d_out, int out_size, void* d_ws, size_t ws_size,
                              hipStream_t stream) {
    (void)in_sizes; (void)n_in; (void)out_size; (void)ws_size;
    const float* x    = (const float*)d_in[0];   // [64,256,32,32]
    const float* cent = (const float*)d_in[1];   // [256,1024]
    const float* csz  = (const float*)d_in[2];   // [1024]
    const float* avg  = (const float*)d_in[3];   // [256,1024]
    float* out = (float*)d_out;

    char* ws = (char*)d_ws;
    __bf16*   cT     = (__bf16*)(ws + 0);              // 1 MB (hi/lo, tiled)
    float*    cnorm  = (float*)(ws + 1048576);         // 4 KB
    int*      ind    = (int*)(ws + 1052672);           // 256 KB
    unsigned* counts = (unsigned*)(ws + 1314816);      // 4 KB
    float*    sums   = (float*)(ws + 1318912);         // 1 MB
    float*    csn    = (float*)(ws + 2367488);         // 4 KB
    float*    newc   = (float*)(ws + 2371584);         // 1 MB

    kPrep  <<<4,     256, 0, stream>>>(cent, cT, cnorm);
    kZero  <<<1,     256, 0, stream>>>(counts);
    kArgmin<<<512,   256, 0, stream>>>(x, cT, cnorm, ind, counts);
    kSums  <<<64,    256, 0, stream>>>(x, ind, sums);
    kNorm  <<<1,     256, 0, stream>>>(csz, counts, csn);
    kNewC  <<<1024,  256, 0, stream>>>(avg, sums, csn, newc);
    kGather<<<16384, 256, 0, stream>>>(newc, ind, out);
}
